// TPShardedExpertNetwork_53695681135019
// MI455X (gfx1250) — matrix operand order repaired and checked by reading, not compiled, not run on hardware
//
#include <hip/hip_runtime.h>

typedef __attribute__((ext_vector_type(16))) __bf16 v16bf;
typedef __attribute__((ext_vector_type(2)))  __bf16 v2bf;
typedef __attribute__((ext_vector_type(8)))  float  v8f;

#define E_ 8
#define T_ 1024
#define D_ 2048
#define I_ 4096

// ---------------------------------------------------------------------------
// helpers
// ---------------------------------------------------------------------------
__device__ __forceinline__ unsigned short f2bf(float f) {
    __bf16 b = (__bf16)f;                      // hw v_cvt bf16 path
    return __builtin_bit_cast(unsigned short, b);
}

__device__ __forceinline__ unsigned int pack2bf(float a, float b) {
    v2bf v = { (__bf16)a, (__bf16)b };         // should select v_cvt_pk_bf16_f32
    return __builtin_bit_cast(unsigned int, v);
}

__device__ __forceinline__ v8f wmma_bf16(v16bf a, v16bf b, v8f c) {
    return __builtin_amdgcn_wmma_f32_16x16x32_bf16(
        /*neg_a=*/false, a, /*neg_b=*/false, b,
        /*c_mod=*/(short)0, c, /*reuse_a=*/false, /*reuse_b=*/false);
}

// A fragment: 16x32 bf16 tile. lane<16 -> row=lane, K {0..7,16..23};
// lane>=16 -> row=lane-16, K {8..15,24..31}. Two 16B LDS loads.
__device__ __forceinline__ v16bf load_fragA(const unsigned short* tile,
                                            int lane, int kb, int stride) {
    const int r   = lane & 15;
    const int off = (lane & 16) >> 1;          // 0 or 8
    const unsigned short* p = tile + r * stride + kb + off;
    union { v16bf v; uint4 u[2]; } f;
    f.u[0] = *(const uint4*)(p);
    f.u[1] = *(const uint4*)(p + 16);
    return f.v;
}

// B fragment from N-major (transposed) LDS tile Bt[n][k].
// lane<16 -> col=lane, K {0..7,16..23}; lane>=16 -> col=lane-16, K {8..15,24..31}.
__device__ __forceinline__ v16bf load_fragB(const unsigned short* tile,
                                            int lane, int kb, int stride) {
    const int n   = lane & 15;
    const int off = (lane & 16) >> 1;          // 0 or 8
    const unsigned short* p = tile + n * stride + kb + off;
    union { v16bf v; uint4 u[2]; } f;
    f.u[0] = *(const uint4*)(p);
    f.u[1] = *(const uint4*)(p + 16);
    return f.v;
}

// ---------------------------------------------------------------------------
// Kernel 1: fused gate/inner GEMMs + SiLU-GLU.  h(bf16) -> workspace
// block tile: 128 (T) x 64 (I); K over D in steps of 64; 8 waves (4x2)
// Double-buffered LDS, register-staged global loads.
// ---------------------------------------------------------------------------
#define K1_BM  128
#define K1_BN  64
#define K1_BK  64
#define K1_BKP 72   // padded row (144B: 16B aligned, bank-skewed)

__global__ __launch_bounds__(256) void glu_gemm12(
    const float* __restrict__ x,
    const float* __restrict__ gate,
    const float* __restrict__ inner,
    unsigned short* __restrict__ hws)
{
    __shared__ unsigned short As[2][K1_BM * K1_BKP];
    __shared__ unsigned short Bg[2][K1_BN * K1_BKP];
    __shared__ unsigned short Bi[2][K1_BN * K1_BKP];

    const int e    = blockIdx.z;
    const int m0   = blockIdx.y * K1_BM;
    const int n0   = blockIdx.x * K1_BN;
    const int tid  = threadIdx.x;
    const int lane = tid & 31;
    const int wid  = tid >> 5;
    const int wm   = wid >> 1;                 // 0..3 : 32-row slab
    const int wn   = wid & 1;                  // 0..1 : 32-col slab

    const v8f zf = {0.f,0.f,0.f,0.f,0.f,0.f,0.f,0.f};
    v8f accG[2][2], accU[2][2];
#pragma unroll
    for (int im = 0; im < 2; ++im)
#pragma unroll
        for (int in = 0; in < 2; ++in) { accG[im][in] = zf; accU[im][in] = zf; }

    const long xbase = ((long)e * T_ + m0) * D_;
    const long wbase = (long)e * D_ * I_ + n0;

    float4 rx[8], rg[4], ri[4];

    // ---- prologue: stage tile k0=0 ----
#pragma unroll
    for (int p = 0; p < 8; ++p) {
        int idx = p * 256 + tid;
        int m = idx >> 4, c = (idx & 15) << 2;
        rx[p] = *(const float4*)(x + xbase + (long)m * D_ + c);
    }
#pragma unroll
    for (int p = 0; p < 4; ++p) {
        int idx = p * 256 + tid;
        int k = idx >> 4, n = (idx & 15) << 2;
        long off = wbase + (long)k * I_ + n;
        rg[p] = *(const float4*)(gate  + off);
        ri[p] = *(const float4*)(inner + off);
    }
#pragma unroll
    for (int p = 0; p < 8; ++p) {
        int idx = p * 256 + tid;
        int m = idx >> 4, c = (idx & 15) << 2;
        *(uint2*)&As[0][m * K1_BKP + c] =
            make_uint2(pack2bf(rx[p].x, rx[p].y), pack2bf(rx[p].z, rx[p].w));
    }
#pragma unroll
    for (int p = 0; p < 4; ++p) {
        int idx = p * 256 + tid;
        int k = idx >> 4, n = (idx & 15) << 2;
        Bg[0][(n + 0) * K1_BKP + k] = f2bf(rg[p].x);
        Bg[0][(n + 1) * K1_BKP + k] = f2bf(rg[p].y);
        Bg[0][(n + 2) * K1_BKP + k] = f2bf(rg[p].z);
        Bg[0][(n + 3) * K1_BKP + k] = f2bf(rg[p].w);
        Bi[0][(n + 0) * K1_BKP + k] = f2bf(ri[p].x);
        Bi[0][(n + 1) * K1_BKP + k] = f2bf(ri[p].y);
        Bi[0][(n + 2) * K1_BKP + k] = f2bf(ri[p].z);
        Bi[0][(n + 3) * K1_BKP + k] = f2bf(ri[p].w);
    }
    __syncthreads();

    int cur = 0;
    for (int k0 = 0; k0 < D_; k0 += K1_BK) {
        const int knext = k0 + K1_BK;
        // ---- issue next tile's global loads (retire during compute) ----
        if (knext < D_) {
#pragma unroll
            for (int p = 0; p < 8; ++p) {
                int idx = p * 256 + tid;
                int m = idx >> 4, c = (idx & 15) << 2;
                rx[p] = *(const float4*)(x + xbase + (long)m * D_ + knext + c);
            }
#pragma unroll
            for (int p = 0; p < 4; ++p) {
                int idx = p * 256 + tid;
                int k = idx >> 4, n = (idx & 15) << 2;
                long off = wbase + (long)(knext + k) * I_ + n;
                rg[p] = *(const float4*)(gate  + off);
                ri[p] = *(const float4*)(inner + off);
            }
        }
        // ---- load ALL fragments for both k-steps, then compute ----
        {
            v16bf a[2][2], bg[2][2], bi[2][2];
#pragma unroll
            for (int ks = 0; ks < 2; ++ks) {
                const int kb = ks * 32;
#pragma unroll
                for (int im = 0; im < 2; ++im)
                    a[ks][im] = load_fragA(&As[cur][(wm * 32 + im * 16) * K1_BKP], lane, kb, K1_BKP);
#pragma unroll
                for (int in = 0; in < 2; ++in) {
                    bg[ks][in] = load_fragB(&Bg[cur][(wn * 32 + in * 16) * K1_BKP], lane, kb, K1_BKP);
                    bi[ks][in] = load_fragB(&Bi[cur][(wn * 32 + in * 16) * K1_BKP], lane, kb, K1_BKP);
                }
            }
#pragma unroll
            for (int ks = 0; ks < 2; ++ks)
#pragma unroll
                for (int im = 0; im < 2; ++im)
#pragma unroll
                    for (int in = 0; in < 2; ++in) {
                        accG[im][in] = wmma_bf16(a[ks][im], bg[ks][in], accG[im][in]);
                        accU[im][in] = wmma_bf16(a[ks][im], bi[ks][in], accU[im][in]);
                    }
        }
        // ---- convert + store next tile into alternate buffer ----
        if (knext < D_) {
            const int nxt = cur ^ 1;
#pragma unroll
            for (int p = 0; p < 8; ++p) {
                int idx = p * 256 + tid;
                int m = idx >> 4, c = (idx & 15) << 2;
                *(uint2*)&As[nxt][m * K1_BKP + c] =
                    make_uint2(pack2bf(rx[p].x, rx[p].y), pack2bf(rx[p].z, rx[p].w));
            }
#pragma unroll
            for (int p = 0; p < 4; ++p) {
                int idx = p * 256 + tid;
                int k = idx >> 4, n = (idx & 15) << 2;
                Bg[nxt][(n + 0) * K1_BKP + k] = f2bf(rg[p].x);
                Bg[nxt][(n + 1) * K1_BKP + k] = f2bf(rg[p].y);
                Bg[nxt][(n + 2) * K1_BKP + k] = f2bf(rg[p].z);
                Bg[nxt][(n + 3) * K1_BKP + k] = f2bf(rg[p].w);
                Bi[nxt][(n + 0) * K1_BKP + k] = f2bf(ri[p].x);
                Bi[nxt][(n + 1) * K1_BKP + k] = f2bf(ri[p].y);
                Bi[nxt][(n + 2) * K1_BKP + k] = f2bf(ri[p].z);
                Bi[nxt][(n + 3) * K1_BKP + k] = f2bf(ri[p].w);
            }
        }
        __syncthreads();
        cur ^= 1;
    }

    // ---- epilogue: h = silu(g) * u = g * sigmoid(g) * u, store bf16 ----
    const int rbase = (lane >> 4) << 3;        // 0 or 8
    const int cl    = lane & 15;
#pragma unroll
    for (int im = 0; im < 2; ++im)
#pragma unroll
        for (int in = 0; in < 2; ++in) {
            const int gm = m0 + wm * 32 + im * 16 + rbase;
            const int gn = n0 + wn * 32 + in * 16 + cl;
#pragma unroll
            for (int v = 0; v < 8; ++v) {
                float g = accG[im][in][v];
                float u = accU[im][in][v];
                float s = __builtin_amdgcn_rcpf(1.0f + __expf(-g));  // sigmoid
                hws[((long)e * T_ + gm + v) * I_ + gn] = f2bf(g * s * u);
            }
        }
}

// ---------------------------------------------------------------------------
// Kernel 2: out = h @ output_proj.  h bf16 from ws, out fp32.
// block tile: 128 (T) x 128 (D); K over I in steps of 64; 8 waves (4x2),
// each wave 32x64 (2x4 WMMA tiles).  Double-buffered LDS.
// ---------------------------------------------------------------------------
#define K2_BM  128
#define K2_BN  128
#define K2_BK  64
#define K2_BKP 72

__global__ __launch_bounds__(256) void glu_gemm3(
    const unsigned short* __restrict__ hws,
    const float* __restrict__ oproj,
    float* __restrict__ out)
{
    __shared__ unsigned short Ah[2][K2_BM * K2_BKP];
    __shared__ unsigned short Bo[2][K2_BN * K2_BKP];

    const int e    = blockIdx.z;
    const int m0   = blockIdx.y * K2_BM;
    const int n0   = blockIdx.x * K2_BN;
    const int tid  = threadIdx.x;
    const int lane = tid & 31;
    const int wid  = tid >> 5;
    const int wm   = wid >> 1;                 // 0..3 : 32-row slab
    const int wn   = wid & 1;                  // 0..1 : 64-col slab

    const v8f zf = {0.f,0.f,0.f,0.f,0.f,0.f,0.f,0.f};
    v8f acc[2][4];
#pragma unroll
    for (int im = 0; im < 2; ++im)
#pragma unroll
        for (int in = 0; in < 4; ++in) acc[im][in] = zf;

    const long hbase = ((long)e * T_ + m0) * I_;
    const long obase = (long)e * I_ * D_ + n0;

    uint4  ra[4];
    float4 rb[8];

    // ---- prologue: stage tile k0=0 ----
#pragma unroll
    for (int p = 0; p < 4; ++p) {
        int idx = p * 256 + tid;
        int m = idx >> 3, c = (idx & 7) << 3;
        ra[p] = *(const uint4*)(hws + hbase + (long)m * I_ + c);
    }
#pragma unroll
    for (int p = 0; p < 8; ++p) {
        int idx = p * 256 + tid;
        int k = idx >> 5, n = (idx & 31) << 2;
        rb[p] = *(const float4*)(oproj + obase + (long)k * D_ + n);
    }
#pragma unroll
    for (int p = 0; p < 4; ++p) {
        int idx = p * 256 + tid;
        int m = idx >> 3, c = (idx & 7) << 3;
        *(uint4*)&Ah[0][m * K2_BKP + c] = ra[p];
    }
#pragma unroll
    for (int p = 0; p < 8; ++p) {
        int idx = p * 256 + tid;
        int k = idx >> 5, n = (idx & 31) << 2;
        Bo[0][(n + 0) * K2_BKP + k] = f2bf(rb[p].x);
        Bo[0][(n + 1) * K2_BKP + k] = f2bf(rb[p].y);
        Bo[0][(n + 2) * K2_BKP + k] = f2bf(rb[p].z);
        Bo[0][(n + 3) * K2_BKP + k] = f2bf(rb[p].w);
    }
    __syncthreads();

    int cur = 0;
    for (int k0 = 0; k0 < I_; k0 += K2_BK) {
        const int knext = k0 + K2_BK;
        // ---- issue next tile's global loads ----
        if (knext < I_) {
#pragma unroll
            for (int p = 0; p < 4; ++p) {
                int idx = p * 256 + tid;
                int m = idx >> 3, c = (idx & 7) << 3;
                ra[p] = *(const uint4*)(hws + hbase + (long)m * I_ + knext + c);
            }
#pragma unroll
            for (int p = 0; p < 8; ++p) {
                int idx = p * 256 + tid;
                int k = idx >> 5, n = (idx & 31) << 2;
                rb[p] = *(const float4*)(oproj + obase + (long)(knext + k) * D_ + n);
            }
        }
        // ---- load ALL fragments for both k-steps, then compute ----
        {
            v16bf a[2][2], b[2][4];
#pragma unroll
            for (int ks = 0; ks < 2; ++ks) {
                const int kb = ks * 32;
#pragma unroll
                for (int im = 0; im < 2; ++im)
                    a[ks][im] = load_fragA(&Ah[cur][(wm * 32 + im * 16) * K2_BKP], lane, kb, K2_BKP);
#pragma unroll
                for (int in = 0; in < 4; ++in)
                    b[ks][in] = load_fragB(&Bo[cur][(wn * 64 + in * 16) * K2_BKP], lane, kb, K2_BKP);
            }
#pragma unroll
            for (int ks = 0; ks < 2; ++ks)
#pragma unroll
                for (int im = 0; im < 2; ++im)
#pragma unroll
                    for (int in = 0; in < 4; ++in)
                        acc[im][in] = wmma_bf16(a[ks][im], b[ks][in], acc[im][in]);
        }
        // ---- store next tile into alternate buffer ----
        if (knext < I_) {
            const int nxt = cur ^ 1;
#pragma unroll
            for (int p = 0; p < 4; ++p) {
                int idx = p * 256 + tid;
                int m = idx >> 3, c = (idx & 7) << 3;
                *(uint4*)&Ah[nxt][m * K2_BKP + c] = ra[p];
            }
#pragma unroll
            for (int p = 0; p < 8; ++p) {
                int idx = p * 256 + tid;
                int k = idx >> 5, n = (idx & 31) << 2;
                Bo[nxt][(n + 0) * K2_BKP + k] = f2bf(rb[p].x);
                Bo[nxt][(n + 1) * K2_BKP + k] = f2bf(rb[p].y);
                Bo[nxt][(n + 2) * K2_BKP + k] = f2bf(rb[p].z);
                Bo[nxt][(n + 3) * K2_BKP + k] = f2bf(rb[p].w);
            }
        }
        __syncthreads();
        cur ^= 1;
    }

    // ---- epilogue: fp32 store ----
    const int rbase = (lane >> 4) << 3;
    const int cl    = lane & 15;
#pragma unroll
    for (int im = 0; im < 2; ++im)
#pragma unroll
        for (int in = 0; in < 4; ++in) {
            const int gm = m0 + wm * 32 + im * 16 + rbase;
            const int gn = n0 + wn * 64 + in * 16 + cl;
#pragma unroll
            for (int v = 0; v < 8; ++v)
                out[((long)e * T_ + gm + v) * D_ + gn] = acc[im][in][v];
        }
}

// ---------------------------------------------------------------------------
extern "C" void kernel_launch(void* const* d_in, const int* in_sizes, int n_in,
                              void* d_out, int out_size, void* d_ws, size_t ws_size,
                              hipStream_t stream) {
    const float* x     = (const float*)d_in[0];
    const float* gate  = (const float*)d_in[1];
    const float* inner = (const float*)d_in[2];
    const float* oproj = (const float*)d_in[3];
    float* out               = (float*)d_out;
    unsigned short* hws      = (unsigned short*)d_ws;   // h: E*T*I bf16 = 64 MB

    dim3 g1(I_ / K1_BN, T_ / K1_BM, E_);   // 64 x 8 x 8
    glu_gemm12<<<g1, 256, 0, stream>>>(x, gate, inner, hws);

    dim3 g2(D_ / K2_BN, T_ / K2_BM, E_);   // 16 x 8 x 8
    glu_gemm3<<<g2, 256, 0, stream>>>(hws, oproj, out);
}
